// Net_83786222011073
// MI455X (gfx1250) — matrix operand order repaired and checked by reading, not compiled, not run on hardware
//
#include <hip/hip_runtime.h>
#include <math.h>

// ---------------------------------------------------------------------------
// GCN (2 layers) + mean-pool + bilinear gate + MLP head for MI455X (gfx1250).
// Edge aggregation via f32 global atomics (L2-resident working set, ~192MB L2);
// dense node GEMMs via v_wmma_f32_16x16x32_f16 (f16 in, f32 accumulate).
// ---------------------------------------------------------------------------

typedef __attribute__((ext_vector_type(16))) _Float16 v16h;
typedef __attribute__((ext_vector_type(8)))  _Float16 v8h;
typedef __attribute__((ext_vector_type(8)))  float    v8f;

#define NN 50000      // nodes  (50000 = 3125 * 16, exact M tiling)
#define NE 1600000    // edges
#define NG 256        // graphs

// ---- workspace layout (bytes) ----
constexpr size_t AL = 256;
constexpr size_t alignUp(size_t x) { return (x + AL - 1) & ~(AL - 1); }

constexpr size_t OFF_MSG1 = 0;                                 // f32 [NN,128]
constexpr size_t SZ_MSG1  = (size_t)NN * 128 * 4;
constexpr size_t OFF_MSG2 = alignUp(OFF_MSG1 + SZ_MSG1);       // f32 [NN,100]
constexpr size_t SZ_MSG2  = (size_t)NN * 100 * 4;
constexpr size_t OFF_DEG  = alignUp(OFF_MSG2 + SZ_MSG2);       // f32 [NN]
constexpr size_t SZ_DEG   = (size_t)NN * 4;
constexpr size_t OFF_HGS  = alignUp(OFF_DEG + SZ_DEG);         // f32 [NG,20]
constexpr size_t SZ_HGS   = (size_t)NG * 20 * 4;
constexpr size_t OFF_CNT  = OFF_HGS + SZ_HGS;                  // f32 [NG]
constexpr size_t SZ_CNT   = (size_t)NG * 4;
constexpr size_t ZERO_END = OFF_CNT + SZ_CNT;                  // zeroed region
constexpr size_t ZERO_N4  = ZERO_END / 16;                     // float4 count

constexpr size_t OFF_AGG1 = alignUp(ZERO_END);                 // f16 [NN,128]
constexpr size_t SZ_AGG1  = (size_t)NN * 128 * 2;
constexpr size_t OFF_H1   = alignUp(OFF_AGG1 + SZ_AGG1);       // f32 [NN,100]
constexpr size_t SZ_H1    = (size_t)NN * 100 * 4;
constexpr size_t OFF_H2   = alignUp(OFF_H1 + SZ_H1);           // f32 [NN,20]
constexpr size_t SZ_H2    = (size_t)NN * 20 * 4;
constexpr size_t OFF_WT1  = alignUp(OFF_H2 + SZ_H2);           // f16 [112,128] col-major W1, padded
constexpr size_t SZ_WT1   = (size_t)112 * 128 * 2;
constexpr size_t OFF_WT2  = alignUp(OFF_WT1 + SZ_WT1);         // f16 [32,128]  col-major W2, padded
constexpr size_t SZ_WT2   = (size_t)32 * 128 * 2;

union AFrag { v16h v; _Float16 h[16]; };

// ---------------------------------------------------------------------------
__global__ void zero_f4(float4* p, int n4) {
    int i = blockIdx.x * 256 + threadIdx.x;
    if (i < n4) p[i] = make_float4(0.f, 0.f, 0.f, 0.f);
}

// transpose + pad weights into f16, column-major (contiguous K per column)
__global__ void wconv_kernel(const float* __restrict__ W1, const float* __restrict__ W2,
                             _Float16* __restrict__ Wt1, _Float16* __restrict__ Wt2) {
    int idx = blockIdx.x * 256 + threadIdx.x;
    if (idx < 112 * 128) {
        int n = idx >> 7, k = idx & 127;
        Wt1[idx] = (n < 100) ? (_Float16)W1[k * 100 + n] : (_Float16)0.f;
    } else {
        int j = idx - 112 * 128;
        if (j < 32 * 128) {
            int n = j >> 7, k = j & 127;
            Wt2[j] = (n < 20 && k < 100) ? (_Float16)W2[k * 20 + n] : (_Float16)0.f;
        }
    }
}

// layer-1 edge scatter: msg1[dst] += feat[src]; deg[dst] += 1
__global__ void scatter1_kernel(const int* __restrict__ src, const int* __restrict__ dst,
                                const float* __restrict__ feat,
                                float* __restrict__ msg1, float* __restrict__ deg) {
    int idx = blockIdx.x * 256 + threadIdx.x;      // NE * 32 threads (4 feats each)
    int e = idx >> 5;
    if (e >= NE) return;
    int f = (idx & 31) * 4;
    int s = src[e], d = dst[e];
    float4 val = *reinterpret_cast<const float4*>(feat + (size_t)s * 128 + f);
    float* p = msg1 + (size_t)d * 128 + f;
    atomicAdd(p + 0, val.x); atomicAdd(p + 1, val.y);
    atomicAdd(p + 2, val.z); atomicAdd(p + 3, val.w);
    if (f == 0) atomicAdd(deg + d, 1.0f);
}

// agg1 = (deg>0 ? msg1/deg : feat)  -> f16
__global__ void agg1_kernel(const float* __restrict__ msg1, const float* __restrict__ deg,
                            const float* __restrict__ feat, _Float16* __restrict__ agg1) {
    int idx = blockIdx.x * 256 + threadIdx.x;
    if (idx >= NN * 128) return;
    int n = idx >> 7;
    float d = deg[n];
    float v = (d > 0.f) ? msg1[idx] / d : feat[idx];
    agg1[idx] = (_Float16)v;
}

// h1 = relu(agg1 @ W1 + b1), WMMA 16x16x32 f16 -> f32, K=128 (4 steps), N tiles = 7
__global__ void gemm1_kernel(const _Float16* __restrict__ agg1, const _Float16* __restrict__ Wt1,
                             const float* __restrict__ b1, float* __restrict__ h1) {
    const int lane = threadIdx.x & 31;
    const int wave = threadIdx.x >> 5;
    const int tile = blockIdx.x * 8 + wave;
    if (tile >= 3125 * 7) return;                  // wave-uniform exit
    const int tm = tile / 7, tn = tile % 7;
    const int m0 = tm * 16, n0 = tn * 16;
    const int row  = m0 + (lane & 15);
    const int col  = n0 + (lane & 15);
    const int kb_a = (lane >> 4) * 16;             // A: lanes 16-31 hold K+16
    const int kb_b = (lane >> 4) * 16;             // B: lanes 16-31 hold K+16
    v8f c = {};
#pragma unroll
    for (int s = 0; s < 4; ++s) {
        const int ka = s * 32 + kb_a;
        v8h a0 = *reinterpret_cast<const v8h*>(agg1 + (size_t)row * 128 + ka);       // K = ka..ka+7
        v8h a1 = *reinterpret_cast<const v8h*>(agg1 + (size_t)row * 128 + ka + 8);   // K = ka+8..ka+15
        AFrag a;
#pragma unroll
        for (int i = 0; i < 8; ++i) { a.h[i] = a0[i]; a.h[8 + i] = a1[i]; }
        v16h b = *reinterpret_cast<const v16h*>(Wt1 + (size_t)col * 128 + s * 32 + kb_b);
        c = __builtin_amdgcn_wmma_f32_16x16x32_f16(false, a.v, false, b, (short)0, c, false, false);
    }
    const int n = col;
    if (n < 100) {
        const float bias = b1[n];
        const int mb = m0 + 8 * (lane >> 4);
#pragma unroll
        for (int v = 0; v < 8; ++v)
            h1[(size_t)(mb + v) * 100 + n] = fmaxf(c[v] + bias, 0.0f);
    }
}

// layer-2 edge scatter: msg2[dst] += h1[src]  (100 feats = 25 float4 per edge)
__global__ void scatter2_kernel(const int* __restrict__ src, const int* __restrict__ dst,
                                const float* __restrict__ h1, float* __restrict__ msg2) {
    int idx = blockIdx.x * 256 + threadIdx.x;      // NE * 25 threads
    int e = idx / 25;
    if (e >= NE) return;
    int f = (idx % 25) * 4;
    int s = src[e], d = dst[e];
    float4 val = *reinterpret_cast<const float4*>(h1 + (size_t)s * 100 + f);
    float* p = msg2 + (size_t)d * 100 + f;
    atomicAdd(p + 0, val.x); atomicAdd(p + 1, val.y);
    atomicAdd(p + 2, val.z); atomicAdd(p + 3, val.w);
}

// h2 = relu(agg2 @ W2 + b2); agg2 built on the fly, K padded 100 -> 128, N tiles = 2
__global__ void gemm2_kernel(const float* __restrict__ msg2, const float* __restrict__ deg,
                             const float* __restrict__ h1, const _Float16* __restrict__ Wt2,
                             const float* __restrict__ b2, float* __restrict__ h2) {
    const int lane = threadIdx.x & 31;
    const int wave = threadIdx.x >> 5;
    const int tile = blockIdx.x * 8 + wave;
    if (tile >= 3125 * 2) return;
    const int tm = tile >> 1, tn = tile & 1;
    const int m0 = tm * 16, n0 = tn * 16;
    const int row  = m0 + (lane & 15);
    const int kb_a = (lane >> 4) * 16;
    const int kb_b = (lane >> 4) * 16;
    const float d    = deg[row];
    const bool  pos  = d > 0.f;
    const float scale = pos ? 1.0f / d : 1.0f;
    const float* srcrow = (pos ? msg2 : h1) + (size_t)row * 100;
    v8f c = {};
#pragma unroll
    for (int s = 0; s < 4; ++s) {
        AFrag a;
#pragma unroll
        for (int i = 0; i < 16; ++i) {
            // K = s*32+kb_a+i
            const int k = s * 32 + kb_a + i;
            a.h[i] = (k < 100) ? (_Float16)(srcrow[k] * scale) : (_Float16)0.f;
        }
        v16h b = *reinterpret_cast<const v16h*>(Wt2 + (size_t)(n0 + (lane & 15)) * 128 + s * 32 + kb_b);
        c = __builtin_amdgcn_wmma_f32_16x16x32_f16(false, a.v, false, b, (short)0, c, false, false);
    }
    const int n = n0 + (lane & 15);
    if (n < 20) {
        const float bias = b2[n];
        const int mb = m0 + 8 * (lane >> 4);
#pragma unroll
        for (int v = 0; v < 8; ++v)
            h2[(size_t)(mb + v) * 20 + n] = fmaxf(c[v] + bias, 0.0f);
    }
}

// per-graph sum pooling (+ node counts)
__global__ void pool_kernel(const float* __restrict__ h2, const int* __restrict__ gid,
                            float* __restrict__ hg_sum, float* __restrict__ cnt) {
    int idx = blockIdx.x * 256 + threadIdx.x;      // NN * 20
    if (idx >= NN * 20) return;
    int n = idx / 20, f = idx % 20;
    int g = gid[n];
    atomicAdd(hg_sum + g * 20 + f, h2[idx]);
    if (f == 0) atomicAdd(cnt + g, 1.0f);
}

// mean pool + bilinear gate + 20->10->1 MLP head; one thread per graph
__global__ void head_kernel(const float* __restrict__ hg_sum, const float* __restrict__ cnt,
                            const float* __restrict__ self_feat,
                            const float* __restrict__ Wp, const float* __restrict__ bp,
                            const float* __restrict__ Wf1, const float* __restrict__ bf1,
                            const float* __restrict__ Wf2, const float* __restrict__ bf2,
                            float* __restrict__ out) {
    const int g = threadIdx.x;                     // blockDim = 256 = NG
    const float inv = 1.0f / fmaxf(cnt[g], 1.0f);
    float fv[20];
#pragma unroll 4
    for (int j = 0; j < 20; ++j) {
        float hg = hg_sum[g * 20 + j] * inv;
        float zp = bp[j];
        for (int k = 0; k < 64; ++k) zp += self_feat[g * 64 + k] * Wp[k * 20 + j];
        float gate = 1.0f / (1.0f + __expf(-hg * zp));
        fv[j] = gate * hg + (1.0f - gate) * zp;
    }
    float acc = bf2[0];
#pragma unroll
    for (int i = 0; i < 10; ++i) {
        float t = bf1[i];
#pragma unroll
        for (int j = 0; j < 20; ++j) t += fv[j] * Wf1[j * 10 + i];
        acc += fmaxf(t, 0.0f) * Wf2[i];
    }
    out[g] = acc;
}

// ---------------------------------------------------------------------------
extern "C" void kernel_launch(void* const* d_in, const int* in_sizes, int n_in,
                              void* d_out, int out_size, void* d_ws, size_t ws_size,
                              hipStream_t stream) {
    (void)in_sizes; (void)n_in; (void)out_size; (void)ws_size;
    const float* feat      = (const float*)d_in[0];
    const int*   src       = (const int*)  d_in[1];
    const int*   dst       = (const int*)  d_in[2];
    const int*   gid       = (const int*)  d_in[3];
    const float* self_feat = (const float*)d_in[4];
    const float* W1  = (const float*)d_in[5];   const float* b1  = (const float*)d_in[6];
    const float* W2  = (const float*)d_in[7];   const float* b2  = (const float*)d_in[8];
    const float* Wp  = (const float*)d_in[9];   const float* bp  = (const float*)d_in[10];
    const float* Wf1 = (const float*)d_in[11];  const float* bf1 = (const float*)d_in[12];
    const float* Wf2 = (const float*)d_in[13];  const float* bf2 = (const float*)d_in[14];
    float* out = (float*)d_out;

    char* ws = (char*)d_ws;
    float*    msg1   = (float*)   (ws + OFF_MSG1);
    float*    msg2   = (float*)   (ws + OFF_MSG2);
    float*    deg    = (float*)   (ws + OFF_DEG);
    float*    hg_sum = (float*)   (ws + OFF_HGS);
    float*    cnt    = (float*)   (ws + OFF_CNT);
    _Float16* agg1   = (_Float16*)(ws + OFF_AGG1);
    float*    h1     = (float*)   (ws + OFF_H1);
    float*    h2     = (float*)   (ws + OFF_H2);
    _Float16* Wt1    = (_Float16*)(ws + OFF_WT1);
    _Float16* Wt2    = (_Float16*)(ws + OFF_WT2);

    // 1. zero accumulators (msg1, msg2, deg, hg_sum, cnt)
    zero_f4<<<(int)((ZERO_N4 + 255) / 256), 256, 0, stream>>>((float4*)ws, (int)ZERO_N4);
    // 2. weight convert (f16, transposed + padded)
    wconv_kernel<<<(112 * 128 + 32 * 128 + 255) / 256, 256, 0, stream>>>(W1, W2, Wt1, Wt2);
    // 3. layer-1 edge scatter + degree
    scatter1_kernel<<<(NE * 32) / 256, 256, 0, stream>>>(src, dst, feat, msg1, deg);
    // 4. aggregate -> f16
    agg1_kernel<<<(NN * 128) / 256, 256, 0, stream>>>(msg1, deg, feat, agg1);
    // 5. layer-1 WMMA GEMM (+bias+relu)
    gemm1_kernel<<<(3125 * 7 + 7) / 8, 256, 0, stream>>>(agg1, Wt1, b1, h1);
    // 6. layer-2 edge scatter
    scatter2_kernel<<<(NE * 25 + 255) / 256, 256, 0, stream>>>(src, dst, h1, msg2);
    // 7. layer-2 WMMA GEMM (+bias+relu)
    gemm2_kernel<<<(3125 * 2 + 7) / 8, 256, 0, stream>>>(msg2, deg, h1, Wt2, b2, h2);
    // 8. per-graph pooling
    pool_kernel<<<(NN * 20 + 255) / 256, 256, 0, stream>>>(h2, gid, hg_sum, cnt);
    // 9. gate + MLP head
    head_kernel<<<1, NG, 0, stream>>>(hg_sum, cnt, self_feat, Wp, bp, Wf1, bf1, Wf2, bf2, out);
}
